// _PulseMoE_45878840656612
// MI455X (gfx1250) — matrix operand order repaired and checked by reading, not compiled, not run on hardware
//
#include <hip/hip_runtime.h>
#include <math.h>

typedef __attribute__((ext_vector_type(16))) __bf16 v16bf;
typedef __attribute__((ext_vector_type(8)))  float  v8f;

#define D_MODEL_ 1024
#define D_FF_    4096
#define NEXP_    8
#define NTOK_    4096
#define NSLOT_   8192
#define SCALE_F  0.894427190999916f   // 1/sqrt(1 + 2/8)

#define BM 128
#define BN 256
#define BK 32
#define LDSS 40   // ushort stride per LDS row (32 + 8 pad), 80B: 16B-aligned, bank-decorrelated

struct __attribute__((aligned(16))) U4 { unsigned int x, y, z, w; };
union Frag16 { v16bf v; U4 q[2]; };

__device__ __forceinline__ unsigned short f2bf(float f) {
  union { float f; unsigned int u; } c; c.f = f;
  unsigned int u = c.u + 0x7FFFu + ((c.u >> 16) & 1u);  // round-to-nearest-even
  return (unsigned short)(u >> 16);
}

__device__ __forceinline__ float gelu_f(float v) {
  const float k = 0.7978845608028654f;  // sqrt(2/pi)
  float t = k * (v + 0.044715f * v * v * v);
  return 0.5f * v * (1.0f + tanhf(t));
}

// ---------------------------------------------------------------------------
// Router: one wave (32 lanes) per token. Lane-parallel dot over D, shfl_xor
// reduce, softmax over 8 experts, top-2 select (lower index wins ties, like
// jax.lax.top_k).
// ---------------------------------------------------------------------------
__global__ __launch_bounds__(256)
void router_kernel(const float* __restrict__ x, const float* __restrict__ wr,
                   float* __restrict__ gates, int* __restrict__ eid)
{
  int tid = threadIdx.x;
  int wave = tid >> 5, lane = tid & 31;
  int token = blockIdx.x * 8 + wave;
  const float4* x4 = (const float4*)(x + (size_t)token * D_MODEL_);
  float4 xr[8];
#pragma unroll
  for (int i = 0; i < 8; i++) xr[i] = x4[i * 32 + lane];
  float logits[NEXP_];
#pragma unroll
  for (int e = 0; e < NEXP_; e++) {
    const float4* w4 = (const float4*)(wr + (size_t)e * D_MODEL_);
    float s = 0.f;
#pragma unroll
    for (int i = 0; i < 8; i++) {
      float4 w = w4[i * 32 + lane];
      s += xr[i].x * w.x + xr[i].y * w.y + xr[i].z * w.z + xr[i].w * w.w;
    }
#pragma unroll
    for (int off = 16; off >= 1; off >>= 1) s += __shfl_xor(s, off, 32);
    logits[e] = s;
  }
  float mx = logits[0];
#pragma unroll
  for (int e = 1; e < NEXP_; e++) mx = fmaxf(mx, logits[e]);
  float p[NEXP_], sum = 0.f;
#pragma unroll
  for (int e = 0; e < NEXP_; e++) { p[e] = __expf(logits[e] - mx); sum += p[e]; }
  float inv = 1.0f / sum;
  int i0 = 0; float v0 = p[0];
#pragma unroll
  for (int e = 1; e < NEXP_; e++) if (p[e] > v0) { v0 = p[e]; i0 = e; }
  int i1 = -1; float v1 = -1.f;
#pragma unroll
  for (int e = 0; e < NEXP_; e++) if (e != i0 && p[e] > v1) { v1 = p[e]; i1 = e; }
  if (lane == 0) {
    gates[token * 2 + 0] = v0 * inv;
    gates[token * 2 + 1] = v1 * inv;
    eid[token * 2 + 0] = i0;
    eid[token * 2 + 1] = i1;
  }
}

// ---------------------------------------------------------------------------
// Build per-expert gather lists. Single block. Row order within an expert is
// atomic-order dependent but the final output is order-invariant (each slot
// scatters to a unique Y row via inv[]).
// ---------------------------------------------------------------------------
__global__ void build_lists_kernel(const int* __restrict__ eid,
                                   int* __restrict__ offsets_g,
                                   int* __restrict__ perm_token,
                                   int* __restrict__ inv)
{
  __shared__ int cnt[NEXP_], offs[NEXP_ + 1], run[NEXP_];
  int tid = threadIdx.x;
  if (tid < NEXP_) cnt[tid] = 0;
  __syncthreads();
  for (int s = tid; s < NSLOT_; s += 256) atomicAdd(&cnt[eid[s]], 1);
  __syncthreads();
  if (tid == 0) {
    offs[0] = 0;
    for (int e = 0; e < NEXP_; e++) offs[e + 1] = offs[e] + cnt[e];
  }
  __syncthreads();
  if (tid < NEXP_) run[tid] = offs[tid];
  __syncthreads();
  for (int s = tid; s < NSLOT_; s += 256) {
    int e = eid[s];
    int pos = atomicAdd(&run[e], 1);
    perm_token[pos] = s >> 1;   // token id
    inv[s] = pos;               // slot -> compacted row
  }
  if (tid <= NEXP_) offsets_g[tid] = offs[tid];
}

// ---------------------------------------------------------------------------
// Unified bf16 WMMA GEMM:  C[m,n] = act( sum_k A[row(m),k] * B[n,k] + bias[n] )
//  - A fp32: converted to bf16 in-register (optionally gathered via row_map)
//  - A bf16: staged with GLOBAL_LOAD_ASYNC_TO_LDS_B128 (ASYNCcnt), no VGPR
//            round trip; invalid boundary rows are EXEC-masked off (stale LDS
//            rows only pollute discarded C rows — row-local contamination)
//  - B: fp32 row-major [N][K] weights, converted during LDS stage, next
//       k-tile prefetched via global_prefetch
//  - offsets != null: blockIdx.z selects expert (weights/bias/M-range)
//  - epilogue: GELU->bf16 store, or fp32 store
// Block: 256 threads = 8 waves in a 2(M)x4(N) grid over a 128x256 tile.
// Each wave owns 64x64 = 4x4 WMMA tiles -> 16 x v_wmma_f32_16x16x32_bf16
// per k-step from 8 fragment loads (0.5 frag-loads/WMMA).
// ---------------------------------------------------------------------------
__global__ __launch_bounds__(256)
void wmma_gemm_kernel(const void* __restrict__ Aptr, int lda, int a_is_bf16,
                      const int* __restrict__ row_map,
                      const float* __restrict__ Bbase, int K,
                      const float* __restrict__ biasbase,
                      void* __restrict__ Cptr, int ldc, int gelu_bf16_out,
                      int Mtotal, int Nsize,
                      const int* __restrict__ offsets)
{
  __shared__ __align__(16) unsigned short lds_a[BM * LDSS];
  __shared__ __align__(16) unsigned short lds_b[BN * LDSS];
  __shared__ int arow_lds[BM];

  int e = blockIdx.z;
  int m0 = 0, Me = Mtotal;
  const float* B = Bbase;
  const float* bias = biasbase;
  if (offsets) {
    m0 = offsets[e];
    Me = offsets[e + 1] - m0;
    B = Bbase + (size_t)e * (size_t)Nsize * (size_t)K;
    bias = biasbase + (size_t)e * Nsize;
  }
  int mtile = blockIdx.y * BM;
  if (mtile >= Me) return;
  int ntile = blockIdx.x * BN;

  int tid = threadIdx.x;
  if (tid < BM) {
    int r = mtile + tid;
    arow_lds[tid] = (r < Me) ? (row_map ? row_map[m0 + r] : (m0 + r)) : -1;
  }

  int wave = tid >> 5, lane = tid & 31;
  int wm = wave >> 2, wn = wave & 3;   // 2 x 4 wave grid
  int lrow = lane & 15;
  int hi = lane >> 4;

  v8f acc[4][4];
  v8f zero = {0.f, 0.f, 0.f, 0.f, 0.f, 0.f, 0.f, 0.f};
#pragma unroll
  for (int i = 0; i < 4; i++)
#pragma unroll
    for (int j = 0; j < 4; j++) acc[i][j] = zero;

  __syncthreads();

  for (int k0 = 0; k0 < K; k0 += BK) {
    // ---- stage A tile (BM x BK) into LDS as bf16 ----
    if (!a_is_bf16) {
      const float* Af = (const float*)Aptr;
#pragma unroll
      for (int i = 0; i < 4; i++) {
        int idx = tid + i * 256;
        int row = idx >> 3, c4 = (idx & 7) * 4;
        int arow = arow_lds[row];
        float4 val = make_float4(0.f, 0.f, 0.f, 0.f);
        if (arow >= 0) val = *(const float4*)(Af + (size_t)arow * lda + k0 + c4);
        unsigned int* dst = (unsigned int*)&lds_a[row * LDSS + c4];
        dst[0] = (unsigned)f2bf(val.x) | ((unsigned)f2bf(val.y) << 16);
        dst[1] = (unsigned)f2bf(val.z) | ((unsigned)f2bf(val.w) << 16);
      }
    } else {
      // Async global->LDS copy (16B/lane), ASYNCcnt-tracked; no conversion
      // needed since A is already bf16.
      const unsigned short* Ah = (const unsigned short*)Aptr;
#pragma unroll
      for (int i = 0; i < 2; i++) {
        int idx = tid + i * 256;
        int row = idx >> 2, c8 = (idx & 3) * 8;
        int arow = arow_lds[row];
        unsigned lds_addr =
            (unsigned)(unsigned long long)(uintptr_t)&lds_a[row * LDSS + c8];
        unsigned long long gaddr =
            (unsigned long long)(uintptr_t)(Ah + (size_t)arow * lda + k0 + c8);
        if (arow >= 0) {
          asm volatile("global_load_async_to_lds_b128 %0, %1, off"
                       :: "v"(lds_addr), "v"(gaddr) : "memory");
        }
      }
    }
    // ---- stage B tile (BN x BK) into LDS as bf16; prefetch next k-tile ----
#pragma unroll
    for (int i = 0; i < 8; i++) {
      int idx = tid + i * 256;
      int row = idx >> 3, c4 = (idx & 7) * 4;
      const float* src = B + (size_t)(ntile + row) * K + k0 + c4;
      float4 val = *(const float4*)src;
      if (k0 + BK < K) __builtin_prefetch((const void*)(src + BK), 0, 3);
      unsigned int* dst = (unsigned int*)&lds_b[row * LDSS + c4];
      dst[0] = (unsigned)f2bf(val.x) | ((unsigned)f2bf(val.y) << 16);
      dst[1] = (unsigned)f2bf(val.z) | ((unsigned)f2bf(val.w) << 16);
    }
    if (a_is_bf16)
      asm volatile("s_wait_asynccnt 0x0" ::: "memory");
    __syncthreads();

    // ---- fragments: contiguous k = 16*hi + s, same order for A and B ----
    Frag16 afr[4], bfr[4];
#pragma unroll
    for (int tm = 0; tm < 4; tm++) {
      int m = wm * 64 + tm * 16 + lrow;
      int kh = hi * 16;  // A: lanes 0-15 K=0..15, lanes 16-31 K=16..31
      afr[tm].q[0] = *(const U4*)&lds_a[m * LDSS + kh];
      afr[tm].q[1] = *(const U4*)&lds_a[m * LDSS + kh + 8];
    }
#pragma unroll
    for (int tn = 0; tn < 4; tn++) {
      int n = wn * 64 + tn * 16 + lrow;
      int kh = hi * 16;  // B: lanes 0-15 K=0..15, lanes 16-31 K=16..31
      bfr[tn].q[0] = *(const U4*)&lds_b[n * LDSS + kh];
      bfr[tn].q[1] = *(const U4*)&lds_b[n * LDSS + kh + 8];
    }
#pragma unroll
    for (int tm = 0; tm < 4; tm++)
#pragma unroll
      for (int tn = 0; tn < 4; tn++)
        acc[tm][tn] = __builtin_amdgcn_wmma_f32_16x16x32_bf16(
            false, afr[tm].v, false, bfr[tn].v, (short)0, acc[tm][tn],
            false, false);
    __syncthreads();
  }

  // ---- epilogue: C layout — lane%16 = N col, VGPR i + 8*(lane>=16) = M row ----
#pragma unroll
  for (int tm = 0; tm < 4; tm++) {
#pragma unroll
    for (int tn = 0; tn < 4; tn++) {
      int ncol = ntile + wn * 64 + tn * 16 + lrow;
      float bv = bias ? bias[ncol] : 0.f;
#pragma unroll
      for (int i = 0; i < 8; i++) {
        int r = wm * 64 + tm * 16 + hi * 8 + i;
        if (mtile + r < Me) {
          size_t crow = (size_t)(m0 + mtile + r);
          float v = acc[tm][tn][i] + bv;
          if (gelu_bf16_out)
            ((unsigned short*)Cptr)[crow * ldc + ncol] = f2bf(gelu_f(v));
          else
            ((float*)Cptr)[crow * ldc + ncol] = v;
        }
      }
    }
  }
}

// ---------------------------------------------------------------------------
// Combine: out = SCALE * (acc_shared + g0 * Y[inv(t,0)] + g1 * Y[inv(t,1)])
// ---------------------------------------------------------------------------
__global__ __launch_bounds__(256)
void combine_kernel(const float* __restrict__ acc, const float* __restrict__ Yr,
                    const float* __restrict__ gates, const int* __restrict__ inv,
                    float* __restrict__ out)
{
  int gid = blockIdx.x * 256 + threadIdx.x;   // float4 index
  int t = gid >> 8;                            // 256 float4 per 1024-float row
  int c = gid & 255;
  float4 a = ((const float4*)acc)[gid];
  int p0 = inv[t * 2], p1 = inv[t * 2 + 1];
  float g0 = gates[t * 2], g1 = gates[t * 2 + 1];
  float4 y0 = ((const float4*)Yr)[(size_t)p0 * 256 + c];
  float4 y1 = ((const float4*)Yr)[(size_t)p1 * 256 + c];
  float4 o;
  o.x = SCALE_F * (a.x + g0 * y0.x + g1 * y1.x);
  o.y = SCALE_F * (a.y + g0 * y0.y + g1 * y1.y);
  o.z = SCALE_F * (a.z + g0 * y0.z + g1 * y1.z);
  o.w = SCALE_F * (a.w + g0 * y0.w + g1 * y1.w);
  ((float4*)out)[gid] = o;
}

// ---------------------------------------------------------------------------
extern "C" void kernel_launch(void* const* d_in, const int* in_sizes, int n_in,
                              void* d_out, int out_size, void* d_ws, size_t ws_size,
                              hipStream_t stream) {
  const float* x    = (const float*)d_in[0];
  const float* wr   = (const float*)d_in[1];
  const float* W1   = (const float*)d_in[2];
  const float* b1   = (const float*)d_in[3];
  const float* W2   = (const float*)d_in[4];
  const float* b2   = (const float*)d_in[5];
  const float* Ws1  = (const float*)d_in[6];
  const float* bs1  = (const float*)d_in[7];
  const float* Ws2  = (const float*)d_in[8];
  const float* bs2  = (const float*)d_in[9];
  float* out = (float*)d_out;

  char* ws = (char*)d_ws;
  size_t off = 0;
  auto alloc = [&](size_t bytes) {
    char* p = ws + off;
    off = (off + bytes + 255) & ~(size_t)255;
    return p;
  };
  float* gates      = (float*)alloc((size_t)NSLOT_ * 4);
  int*   eid        = (int*)  alloc((size_t)NSLOT_ * 4);
  int*   offsets    = (int*)  alloc(16 * 4);
  int*   perm_token = (int*)  alloc((size_t)NSLOT_ * 4);
  int*   inv        = (int*)  alloc((size_t)NSLOT_ * 4);
  float* accbuf     = (float*)alloc((size_t)NTOK_ * D_MODEL_ * 4);        // 16 MB
  unsigned short* Hs = (unsigned short*)alloc((size_t)NTOK_ * D_FF_ * 2); // 32 MB
  float* Yr = (float*)Hs;  // alias: Hs dead after shared FFN2; Yr born after
  unsigned short* Hr = (unsigned short*)alloc((size_t)NSLOT_ * D_FF_ * 2); // 64 MB

  // 1) Router (gates + expert ids per slot)
  router_kernel<<<NTOK_ / 8, 256, 0, stream>>>(x, wr, gates, eid);
  // 2) Per-expert gather lists
  build_lists_kernel<<<1, 256, 0, stream>>>(eid, offsets, perm_token, inv);
  // 3) Shared FFN1: Hs = gelu(x @ Ws1^T + bs1)   [4096 x 4096] bf16
  wmma_gemm_kernel<<<dim3(D_FF_ / BN, NTOK_ / BM, 1), 256, 0, stream>>>(
      x, D_MODEL_, 0, nullptr, Ws1, D_MODEL_, bs1,
      Hs, D_FF_, 1, NTOK_, D_FF_, nullptr);
  // 4) Shared FFN2: acc = Hs @ Ws2^T + bs2       [4096 x 1024] f32
  wmma_gemm_kernel<<<dim3(D_MODEL_ / BN, NTOK_ / BM, 1), 256, 0, stream>>>(
      Hs, D_FF_, 1, nullptr, Ws2, D_FF_, bs2,
      accbuf, D_MODEL_, 0, NTOK_, D_MODEL_, nullptr);
  // 5) Routed FFN1: Hr = gelu(gather(x) @ W1[e]^T + b1[e])  [8192 x 4096] bf16
  wmma_gemm_kernel<<<dim3(D_FF_ / BN, NSLOT_ / BM, NEXP_), 256, 0, stream>>>(
      x, D_MODEL_, 0, perm_token, W1, D_MODEL_, b1,
      Hr, D_FF_, 1, NSLOT_, D_FF_, offsets);
  // 6) Routed FFN2: Yr = Hr @ W2[e]^T + b2[e]    [8192 x 1024] f32
  wmma_gemm_kernel<<<dim3(D_MODEL_ / BN, NSLOT_ / BM, NEXP_), 256, 0, stream>>>(
      Hr, D_FF_, 1, nullptr, W2, D_FF_, b2,
      Yr, D_MODEL_, 0, NSLOT_, D_MODEL_, offsets);
  // 7) Combine with gates + global scale
  combine_kernel<<<(NTOK_ * D_MODEL_ / 4) / 256, 256, 0, stream>>>(
      accbuf, Yr, gates, inv, out);
}
